// MultiHeadAttention_51556787421382
// MI455X (gfx1250) — matrix operand order repaired and checked by reading, not compiled, not run on hardware
//
#include <hip/hip_runtime.h>
#include <hip/hip_bf16.h>

// ---------------------------------------------------------------------------
// MultiHeadAttention for MI455X (gfx1250): flash-attention with f16 WMMA and
// async global->LDS staging of K/V tiles (CDNA5 ASYNCcnt path).
// B=4, S=2048, E=1024, H=16, D=64.
// ---------------------------------------------------------------------------

constexpr int Bd = 4;
constexpr int Sd = 2048;
constexpr int Ed = 1024;
constexpr int Hd = 16;
constexpr int Dd = 64;

typedef __attribute__((ext_vector_type(16))) _Float16 v16h;
typedef __attribute__((ext_vector_type(8)))  _Float16 v8h;
typedef __attribute__((ext_vector_type(8)))  float    v8f;
typedef __attribute__((ext_vector_type(4)))  float    v4f;

// D = A*B + C, f16 inputs, f32 accumulate (codegen-verified builtin).
static __device__ __forceinline__ v8f wmma_f16(v16h a, v16h b, v8f c) {
  return __builtin_amdgcn_wmma_f32_16x16x32_f16(
      /*neg_a=*/false, a, /*neg_b=*/false, b,
      /*c_mod=*/(short)0, c, /*reuse_a=*/false, /*reuse_b=*/false);
}

// Async 16B-per-lane copy global -> LDS (VGLOBAL GLOBAL_LOAD_ASYNC_TO_LDS_B128,
// tracked by ASYNCcnt). ldsaddr = 32-bit LDS byte address (low 32 bits of the
// generic shared pointer, per the aperture truncation rule).
static __device__ __forceinline__ void async_copy_b128(unsigned ldsaddr,
                                                       const void* gptr) {
  asm volatile("global_load_async_to_lds_b128 %0, %1, off"
               :: "v"(ldsaddr), "v"((unsigned long long)gptr)
               : "memory");
}

// A-matrix 16x32 f16 fragment (ISA 7.12.2): lanes 0-15 row=lane, K={0..7,16..23};
// lanes 16-31 row=lane-16, K={8..15,24..31}. Source row-major, k contiguous.
static __device__ __forceinline__ v16h load_a16(const _Float16* base, int stride,
                                                int row0, int k0) {
  const int lane = threadIdx.x & 31;
  const int hp = lane >> 4;
  const int r  = lane & 15;
  const _Float16* p = base + (size_t)(row0 + r) * stride + k0 + hp * 8;
  v8h lo = *(const v8h*)p;
  v8h hi = *(const v8h*)(p + 16);
  v16h out;
#pragma unroll
  for (int i = 0; i < 8; ++i) { out[i] = lo[i]; out[i + 8] = hi[i]; }
  return out;
}

// Same A layout but converting from f32 source on the fly.
static __device__ __forceinline__ v16h load_a32f(const float* base, int stride,
                                                 int row0, int k0) {
  const int lane = threadIdx.x & 31;
  const int hp = lane >> 4;
  const int r  = lane & 15;
  const float* p = base + (size_t)(row0 + r) * stride + k0 + hp * 8;
  v4f a = *(const v4f*)p;
  v4f b = *(const v4f*)(p + 4);
  v4f c = *(const v4f*)(p + 16);
  v4f d = *(const v4f*)(p + 20);
  v16h out;
#pragma unroll
  for (int i = 0; i < 4; ++i) {
    out[i]      = (_Float16)a[i];
    out[i + 4]  = (_Float16)b[i];
    out[i + 8]  = (_Float16)c[i];
    out[i + 12] = (_Float16)d[i];
  }
  return out;
}

// B-matrix 32x16 f16 fragment: lanes 0-15 col=lane K={0..7,16..23}; lanes 16-31
// col=lane-16 K={8..15,24..31} (same k order as A). Memory
// is laid out so that for fixed col, k is contiguous -> two aligned 16B loads.
static __device__ __forceinline__ v16h load_b16(const _Float16* base, int stride,
                                                int col0, int k0) {
  const int lane = threadIdx.x & 31;
  const int hp = lane >> 4;
  const int r  = lane & 15;
  const _Float16* p = base + (size_t)(col0 + r) * stride + k0 + hp * 8;
  v8h lo = *(const v8h*)p;
  v8h hi = *(const v8h*)(p + 16);
  v16h out;
#pragma unroll
  for (int i = 0; i < 8; ++i) { out[i] = lo[i]; out[i + 8] = hi[i]; }
  return out;
}

// ---------------------------------------------------------------------------
// Kernel 0: weight prep. WqT/WkT/WvT[h][e][d] = W[h][d][e] (f16), Wo -> f16.
// ---------------------------------------------------------------------------
__global__ void __launch_bounds__(256) prep_weights_kernel(
    const float* __restrict__ Wq, const float* __restrict__ Wk,
    const float* __restrict__ Wv, const float* __restrict__ Wo,
    _Float16* __restrict__ wqT, _Float16* __restrict__ wkT,
    _Float16* __restrict__ wvT, _Float16* __restrict__ woH) {
  const int tid = blockIdx.x * blockDim.x + threadIdx.x;
  const int nth = gridDim.x * blockDim.x;
  const int nW = Hd * Dd * Dd;           // 65536
  for (int i = tid; i < nW; i += nth) {
    const int h = i >> 12;
    const int e = (i >> 6) & 63;
    const int d = i & 63;
    const int src = (h << 12) + (d << 6) + e;
    wqT[i] = (_Float16)Wq[src];
    wkT[i] = (_Float16)Wk[src];
    wvT[i] = (_Float16)Wv[src];
  }
  const int nO = Ed * Ed;
  for (int i = tid; i < nO; i += nth) woH[i] = (_Float16)Wo[i];
}

// ---------------------------------------------------------------------------
// Kernel 1: QKV projection. One wave per (b,h, 16-row s-tile) -> 16x64 tile
// per matrix. Q,K stored [b][h][s][d] f16; V stored transposed [b][h][d][s].
// ---------------------------------------------------------------------------
__global__ void __launch_bounds__(256) qkv_proj_kernel(
    const float* __restrict__ x,
    const float* __restrict__ bq, const float* __restrict__ bk,
    const float* __restrict__ bv,
    const _Float16* __restrict__ wqT, const _Float16* __restrict__ wkT,
    const _Float16* __restrict__ wvT,
    _Float16* __restrict__ Qb, _Float16* __restrict__ Kb,
    _Float16* __restrict__ Vt) {
  const int wave = threadIdx.x >> 5;
  const int lane = threadIdx.x & 31;
  const int hp = lane >> 4;
  const int cl = lane & 15;
  const int t  = blockIdx.x * 8 + wave;   // 8192 tiles total
  const int st = t & 127;                 // S/16 tiles per (b,h)
  const int bh = t >> 7;                  // b*H + h
  const int b  = bh >> 4;
  const int h  = bh & 15;
  const int srow = st * 16;

  // A fragments from x (head slice), shared by Q/K/V.
  const float* xbase = x + (size_t)b * Sd * Ed + (size_t)h * Dd;
  v16h a0 = load_a32f(xbase, Ed, srow, 0);
  v16h a1 = load_a32f(xbase, Ed, srow, 32);

  const _Float16* wq = wqT + (size_t)h * Dd * Dd;
  const _Float16* wk = wkT + (size_t)h * Dd * Dd;
  const _Float16* wv = wvT + (size_t)h * Dd * Dd;

#pragma unroll
  for (int nt = 0; nt < 4; ++nt) {
    const int n = nt * 16 + cl;           // output channel this lane owns
    // ---- Q ----
    {
      v8f c = {};
      c = wmma_f16(a0, load_b16(wq, Dd, nt * 16, 0), c);
      c = wmma_f16(a1, load_b16(wq, Dd, nt * 16, 32), c);
      const float bias = bq[h * Dd + n];
      _Float16* dst = Qb + ((size_t)bh * Sd + srow + hp * 8) * Dd + n;
#pragma unroll
      for (int r = 0; r < 8; ++r) dst[(size_t)r * Dd] = (_Float16)(c[r] + bias);
    }
    // ---- K ----
    {
      v8f c = {};
      c = wmma_f16(a0, load_b16(wk, Dd, nt * 16, 0), c);
      c = wmma_f16(a1, load_b16(wk, Dd, nt * 16, 32), c);
      const float bias = bk[h * Dd + n];
      _Float16* dst = Kb + ((size_t)bh * Sd + srow + hp * 8) * Dd + n;
#pragma unroll
      for (int r = 0; r < 8; ++r) dst[(size_t)r * Dd] = (_Float16)(c[r] + bias);
    }
    // ---- V (store transposed [d][s], rows contiguous in s -> packed 16B) ----
    {
      v8f c = {};
      c = wmma_f16(a0, load_b16(wv, Dd, nt * 16, 0), c);
      c = wmma_f16(a1, load_b16(wv, Dd, nt * 16, 32), c);
      const float bias = bv[h * Dd + n];
      v8h pk;
#pragma unroll
      for (int r = 0; r < 8; ++r) pk[r] = (_Float16)(c[r] + bias);
      *(v8h*)(Vt + ((size_t)bh * Dd + n) * Sd + srow + hp * 8) = pk;
    }
  }
}

// ---------------------------------------------------------------------------
// Kernel 2: flash attention. One wave per 16 query rows; key blocks of 32.
// K/V tiles are staged into LDS once per workgroup with double-buffered
// GLOBAL_LOAD_ASYNC_TO_LDS_B128 (ASYNCcnt), overlapping the copy of block
// i+1 with the WMMA/softmax of block i. Scores stay in registers; online
// softmax uses 16-lane shuffle reductions matching the C-tile striping; P is
// bounced through per-wave LDS to convert the C layout into the A layout.
// ---------------------------------------------------------------------------
__global__ void __launch_bounds__(256) attention_kernel(
    const _Float16* __restrict__ Qb, const _Float16* __restrict__ Kb,
    const _Float16* __restrict__ Vt, _Float16* __restrict__ AO) {
  __shared__ __align__(16) _Float16 ldsK[2][32][64];   // 2 x 4 KB key tiles
  __shared__ __align__(16) _Float16 ldsV[2][64][32];   // 2 x 4 KB value tiles
  __shared__ __align__(32) _Float16 pbuf[8][16][32];   // per-wave P transpose

  const int wave = threadIdx.x >> 5;
  const int lane = threadIdx.x & 31;
  const int t256 = threadIdx.x;
  const int hp = lane >> 4;
  const int cl = lane & 15;
  const int bh = blockIdx.x >> 4;         // b*H + h (64)
  const int qb = blockIdx.x & 15;         // 16 q-blocks of 128 rows
  const int b  = bh >> 4;
  const int h  = bh & 15;
  const int qrow = qb * 128 + wave * 16;

  const _Float16* Qh = Qb + (size_t)bh * Sd * Dd;
  const _Float16* Kh = Kb + (size_t)bh * Sd * Dd;
  const _Float16* Vh = Vt + (size_t)bh * Dd * Sd;

  // Stage one 32-key block (K: 32x64 f16, V: 64x32 f16) into LDS buffer `buf`.
  // 256 threads x 16 B = 4 KB per tile; each wave contributes 2 async ops.
  auto stage = [&](int buf, int j0) {
    const _Float16* gK = Kh + (size_t)j0 * Dd + t256 * 8;
    const unsigned lK =
        (unsigned)(size_t)(&ldsK[buf][0][0]) + (unsigned)(t256 * 16);
    async_copy_b128(lK, gK);
    const _Float16* gV = Vh + (size_t)(t256 >> 2) * Sd + j0 + (t256 & 3) * 8;
    const unsigned lV = (unsigned)(size_t)(&ldsV[buf][0][0]) +
                        (unsigned)((t256 >> 2) * 64 + (t256 & 3) * 16);
    async_copy_b128(lV, gV);
  };

  v16h aq0 = load_a16(Qh, Dd, qrow, 0);
  v16h aq1 = load_a16(Qh, Dd, qrow, 32);

  float rowm[8], rowl[8];
  v8f o0 = {}, o1 = {}, o2 = {}, o3 = {};
#pragma unroll
  for (int r = 0; r < 8; ++r) { rowm[r] = -3.0e38f; rowl[r] = 0.0f; }

  const float sc = 0.125f;                // 1/sqrt(64)

  stage(0, 0);                            // prologue fill of buffer 0

  for (int it = 0; it < Sd / 32; ++it) {
    const int j0  = it * 32;
    const int cur = it & 1;

    if (j0 + 32 < Sd) {
      stage(cur ^ 1, j0 + 32);            // overlap next copy with compute
      asm volatile("s_wait_asynccnt 0x2" ::: "memory");  // cur buffer done
    } else {
      asm volatile("s_wait_asynccnt 0x0" ::: "memory");
    }
    __syncthreads();                      // cur tile visible to all waves

    const _Float16* Kt = &ldsK[cur][0][0];
    const _Float16* Vtl = &ldsV[cur][0][0];

    // ---- scores: 16x32 tile of Q.K^T from LDS ----
    v8f c0 = {}, c1 = {};
    c0 = wmma_f16(aq0, load_b16(Kt, Dd, 0, 0), c0);
    c0 = wmma_f16(aq1, load_b16(Kt, Dd, 0, 32), c0);
    c1 = wmma_f16(aq0, load_b16(Kt, Dd, 16, 0), c1);
    c1 = wmma_f16(aq1, load_b16(Kt, Dd, 16, 32), c1);

    // ---- online softmax (row stats replicated across each 16-lane half) ----
    float p0[8], p1[8];
#pragma unroll
    for (int r = 0; r < 8; ++r) {
      const float s0 = c0[r] * sc;
      const float s1 = c1[r] * sc;
      float m = fmaxf(s0, s1);
      m = fmaxf(m, __shfl_xor(m, 1, 32));
      m = fmaxf(m, __shfl_xor(m, 2, 32));
      m = fmaxf(m, __shfl_xor(m, 4, 32));
      m = fmaxf(m, __shfl_xor(m, 8, 32));
      const float mn = fmaxf(rowm[r], m);
      const float al = __expf(rowm[r] - mn);
      const float e0 = __expf(s0 - mn);
      const float e1 = __expf(s1 - mn);
      float rs = e0 + e1;
      rs += __shfl_xor(rs, 1, 32);
      rs += __shfl_xor(rs, 2, 32);
      rs += __shfl_xor(rs, 4, 32);
      rs += __shfl_xor(rs, 8, 32);
      rowl[r] = rowl[r] * al + rs;
      rowm[r] = mn;
      p0[r] = e0;
      p1[r] = e1;
      o0[r] *= al; o1[r] *= al; o2[r] *= al; o3[r] *= al;
    }

    // ---- C-layout -> A-layout transpose of P through per-wave LDS ----
#pragma unroll
    for (int r = 0; r < 8; ++r) {
      pbuf[wave][hp * 8 + r][cl]      = (_Float16)p0[r];
      pbuf[wave][hp * 8 + r][16 + cl] = (_Float16)p1[r];
    }
    asm volatile("s_wait_dscnt 0" ::: "memory");   // wave-local LDS RAW fence
    v16h ap = load_a16(&pbuf[wave][0][0], 32, 0, 0);

    // ---- O += P.V from the staged (pre-transposed) V tile ----
    o0 = wmma_f16(ap, load_b16(Vtl, 32, 0,  0), o0);
    o1 = wmma_f16(ap, load_b16(Vtl, 32, 16, 0), o1);
    o2 = wmma_f16(ap, load_b16(Vtl, 32, 32, 0), o2);
    o3 = wmma_f16(ap, load_b16(Vtl, 32, 48, 0), o3);

    __syncthreads();   // all waves done reading cur before it is re-filled
  }

  // ---- normalize and write concatenated-head output (f16) ----
  _Float16* dst = AO + ((size_t)b * Sd + qrow + hp * 8) * Ed + h * Dd;
#pragma unroll
  for (int r = 0; r < 8; ++r) {
    const float inv = 1.0f / rowl[r];
    _Float16* drow = dst + (size_t)r * Ed;
    drow[cl]      = (_Float16)(o0[r] * inv);
    drow[16 + cl] = (_Float16)(o1[r] * inv);
    drow[32 + cl] = (_Float16)(o2[r] * inv);
    drow[48 + cl] = (_Float16)(o3[r] * inv);
  }
}

// ---------------------------------------------------------------------------
// Kernel 3: output projection out = AO @ Wo^T + bo (f32 result).
// One wave per 16x64 output tile, 32 K-chunks over E=1024.
// ---------------------------------------------------------------------------
__global__ void __launch_bounds__(256) out_proj_kernel(
    const _Float16* __restrict__ AO, const _Float16* __restrict__ woH,
    const float* __restrict__ bo, float* __restrict__ out) {
  const int wave = threadIdx.x >> 5;
  const int lane = threadIdx.x & 31;
  const int hp = lane >> 4;
  const int cl = lane & 15;
  const int t  = blockIdx.x * 8 + wave;   // 8192 tiles
  const int ng = t & 15;                  // 16 column groups of 64
  const int rt = t >> 4;                  // 512 row tiles over B*S
  const int row0 = rt * 16;

  v8f acc0 = {}, acc1 = {}, acc2 = {}, acc3 = {};
  for (int kc = 0; kc < Ed / 32; ++kc) {
    const int k0 = kc * 32;
    v16h a = load_a16(AO, Ed, row0, k0);
    acc0 = wmma_f16(a, load_b16(woH, Ed, ng * 64 + 0,  k0), acc0);
    acc1 = wmma_f16(a, load_b16(woH, Ed, ng * 64 + 16, k0), acc1);
    acc2 = wmma_f16(a, load_b16(woH, Ed, ng * 64 + 32, k0), acc2);
    acc3 = wmma_f16(a, load_b16(woH, Ed, ng * 64 + 48, k0), acc3);
  }
  v8f accs[4] = {acc0, acc1, acc2, acc3};
#pragma unroll
  for (int nt = 0; nt < 4; ++nt) {
    const int n = ng * 64 + nt * 16 + cl;
    const float bias = bo[n];
    float* dst = out + (size_t)(row0 + hp * 8) * Ed + n;
#pragma unroll
    for (int r = 0; r < 8; ++r) dst[(size_t)r * Ed] = accs[nt][r] + bias;
  }
}

// ---------------------------------------------------------------------------
extern "C" void kernel_launch(void* const* d_in, const int* in_sizes, int n_in,
                              void* d_out, int out_size, void* d_ws,
                              size_t ws_size, hipStream_t stream) {
  const float* x  = (const float*)d_in[0];
  const float* Wq = (const float*)d_in[1];
  const float* Wk = (const float*)d_in[2];
  const float* Wv = (const float*)d_in[3];
  const float* bq = (const float*)d_in[4];
  const float* bk = (const float*)d_in[5];
  const float* bv = (const float*)d_in[6];
  const float* Wo = (const float*)d_in[7];
  const float* bo = (const float*)d_in[8];
  float* out = (float*)d_out;

  // Workspace carve-up (all f16, all offsets 256B-aligned by construction).
  char* ws = (char*)d_ws;
  _Float16* wqT = (_Float16*)ws; ws += (size_t)Hd * Dd * Dd * 2;      // 128 KB
  _Float16* wkT = (_Float16*)ws; ws += (size_t)Hd * Dd * Dd * 2;
  _Float16* wvT = (_Float16*)ws; ws += (size_t)Hd * Dd * Dd * 2;
  _Float16* woH = (_Float16*)ws; ws += (size_t)Ed * Ed * 2;           // 2 MB
  _Float16* Qb  = (_Float16*)ws; ws += (size_t)Bd * Hd * Sd * Dd * 2; // 16 MB
  _Float16* Kb  = (_Float16*)ws; ws += (size_t)Bd * Hd * Sd * Dd * 2; // 16 MB
  _Float16* Vt  = (_Float16*)ws; ws += (size_t)Bd * Hd * Dd * Sd * 2; // 16 MB
  _Float16* AO  = (_Float16*)ws; ws += (size_t)Bd * Sd * Ed * 2;      // 16 MB

  prep_weights_kernel<<<512, 256, 0, stream>>>(Wq, Wk, Wv, Wo,
                                               wqT, wkT, wvT, woH);
  qkv_proj_kernel<<<1024, 256, 0, stream>>>(x, bq, bk, bv, wqT, wkT, wvT,
                                            Qb, Kb, Vt);
  attention_kernel<<<1024, 256, 0, stream>>>(Qb, Kb, Vt, AO);
  out_proj_kernel<<<1024, 256, 0, stream>>>(AO, woH, bo, out);
}
